// AttentionSink_9637906612576
// MI455X (gfx1250) — matrix operand order repaired and checked by reading, not compiled, not run on hardware
//
#include <hip/hip_runtime.h>
#include <hip/hip_bf16.h>

#define NH 32
#define SQ 2048
#define SK 2048
#define DH 128
#define NCHUNK (SK / 32)

typedef __attribute__((ext_vector_type(16))) __bf16 v16bf;
typedef __attribute__((ext_vector_type(8)))  float  v8f;
typedef __attribute__((ext_vector_type(4)))  float  v4f;   // native vector: OK for nontemporal builtins

#define LOG2E 1.44269504088896340736f

// LDS: V chunk transposed, [DH columns][32 K values] in bf16, padded row stride.
// 80 B/row keeps b128 fragment reads 16B-aligned and conflict-free (20-dword bank stride).
#define LDS_STRIDE 40   // halfs (32 used + 8 pad)

__device__ __forceinline__ unsigned short f2bf(float x) {
  union { __bf16 b; unsigned short u; } cv;
  cv.b = (__bf16)x;   // hardware RNE convert
  return cv.u;
}

__launch_bounds__(128, 1)
__global__ void attn_sink_pv(const float* __restrict__ logits,
                             const float* __restrict__ value,
                             const float* __restrict__ sinks,
                             float* __restrict__ out) {
  // double-buffered V tile: 2 x 10,240 B
  __shared__ __align__(16) unsigned short vt[2][DH * LDS_STRIDE];

  const int h    = blockIdx.x >> 5;          // 32 q-blocks per head
  const int qb   = blockIdx.x & 31;
  const int q0   = qb * 64;                  // 4 waves x 16 rows
  const int wave = threadIdx.x >> 5;
  const int lane = threadIdx.x & 31;
  const int r    = lane & 15;                // row (A side) / col-in-tile (B,C side)
  const int hi   = lane >> 4;                // half-wave select
  const int koff = hi * 8;                   // A-fragment K base per ISA layout
  const int kbB  = hi * 8;                   // B-fragment K base per ISA layout
  const int row  = q0 + wave * 16 + r;       // this lane's query row

  const float* lrow = logits + ((size_t)(h * SQ + row)) * SK;
  const float* vbase = value + ((size_t)h * SK) * DH;

  // V staging map: thread -> 4 columns, 4 (even-k, k+1) row pairs
  const int c4 = (threadIdx.x & 31) * 4;
  const int kp = (threadIdx.x >> 5) * 2;

  // ---- helpers ----
  auto loadLogits = [&](int k0, float* x) {
    const v4f* l0 = (const v4f*)(lrow + k0 + koff);
    const v4f* l2 = (const v4f*)(lrow + k0 + koff + 16);
    v4f x0 = __builtin_nontemporal_load(l0);
    v4f x1 = __builtin_nontemporal_load(l0 + 1);
    v4f x2 = __builtin_nontemporal_load(l2);
    v4f x3 = __builtin_nontemporal_load(l2 + 1);
#pragma unroll
    for (int i = 0; i < 4; ++i) {
      x[i]      = x0[i];
      x[i + 4]  = x1[i];
      x[i + 8]  = x2[i];
      x[i + 12] = x3[i];
    }
  };

  auto stageV = [&](int k0, int buf) {
#pragma unroll
    for (int kk = 0; kk < 4; ++kk) {
      const int k = kp + kk * 8;  // even k in [0,32)
      const v4f a = *(const v4f*)(vbase + ((size_t)(k0 + k)) * DH + c4);
      const v4f b = *(const v4f*)(vbase + ((size_t)(k0 + k + 1)) * DH + c4);
#pragma unroll
      for (int j = 0; j < 4; ++j) {
        unsigned int pk = (unsigned int)f2bf(a[j]) | ((unsigned int)f2bf(b[j]) << 16);
        *(unsigned int*)&vt[buf][(c4 + j) * LDS_STRIDE + k] = pk;  // 4B aligned (k even)
      }
    }
  };

  // Online softmax state (log2 domain). Sink folds in as m=sink*log2e, d=1.
  float m = sinks[h] * LOG2E;
  float d = 1.0f;

  v8f acc[8];
#pragma unroll
  for (int t = 0; t < 8; ++t)
#pragma unroll
    for (int e = 0; e < 8; ++e) acc[t][e] = 0.0f;

  // ---- prologue: prime pipeline ----
  float x[16];
  loadLogits(0, x);
  stageV(0, 0);
  __syncthreads();

  for (int c = 0; c < NCHUNK; ++c) {
    const int buf = c & 1;

    // ---------- issue next chunk's memory work first (overlaps with compute) ----------
    float xn[16];
    const bool more = (c + 1 < NCHUNK);   // uniform branch: EXEC stays full
    if (more) {
      loadLogits((c + 1) * 32, xn);
      stageV((c + 1) * 32, buf ^ 1);
    }

    // ---------- online softmax on current chunk (raw logits in x) ----------
    float cm = x[0];
#pragma unroll
    for (int i = 1; i < 16; ++i) cm = fmaxf(cm, x[i]);
    cm = fmaxf(cm, __shfl_xor(cm, 16, 32));    // fold the two half-rows
    const float nm    = fmaxf(m, cm * LOG2E);
    const float scale = exp2f(m - nm);
    m = nm;

    float p[16];
    float s = 0.0f;
#pragma unroll
    for (int i = 0; i < 16; ++i) {
      p[i] = exp2f(__builtin_fmaf(x[i], LOG2E, -nm));
      s += p[i];
    }
    s += __shfl_xor(s, 16, 32);
    d = d * scale + s;

    // ---------- rescale accumulators only when some row max moved ----------
    if (__any(scale < 1.0f)) {                 // wave-uniform -> EXEC stays full
      float f[8];
#pragma unroll
      for (int rr = 0; rr < 8; ++rr) f[rr] = __shfl(scale, rr + hi * 8, 32);
#pragma unroll
      for (int t = 0; t < 8; ++t)
#pragma unroll
        for (int rr = 0; rr < 8; ++rr) acc[t][rr] *= f[rr];
    }

    // ---------- A fragment (bf16), ISA 16-bit 16x32 layout ----------
    v16bf afrag;
#pragma unroll
    for (int i = 0; i < 16; ++i) afrag[i] = (__bf16)p[i];

    // ---------- 8 WMMAs over DH=128, B fragments from current LDS buffer ----------
#pragma unroll
    for (int t = 0; t < 8; ++t) {
      const int cc = t * 16 + r;   // V column for this lane's B fragment
      union { uint4 u[2]; v16bf v; } bf;
      const uint4* bp = (const uint4*)&vt[buf][cc * LDS_STRIDE + kbB];  // 16B aligned
      bf.u[0] = bp[0];
      bf.u[1] = bp[2];
      acc[t] = __builtin_amdgcn_wmma_f32_16x16x32_bf16(
          false, afrag, false, bf.v, (short)0, acc[t], false, false);
    }

    // single barrier per chunk: protects buf^1 writes (this iter) before its reads
    // (next iter) AND buf reads (this iter) before its overwrite (next iter)
    __syncthreads();

    if (more) {
#pragma unroll
      for (int i = 0; i < 16; ++i) x[i] = xn[i];
    }
  }

  // ---------- epilogue: normalize by denom (incl. sink term) and store ----------
  float dv[8];
#pragma unroll
  for (int rr = 0; rr < 8; ++rr) dv[rr] = 1.0f / __shfl(d, rr + hi * 8, 32);

#pragma unroll
  for (int t = 0; t < 8; ++t) {
#pragma unroll
    for (int rr = 0; rr < 8; ++rr) {
      const int M = rr + hi * 8;                       // C/D layout row
      const int q = q0 + wave * 16 + M;
      float* op = out + ((size_t)(h * SQ + q)) * DH + t * 16 + r;
      __builtin_nontemporal_store(acc[t][rr] * dv[rr], op);
    }
  }
}

extern "C" void kernel_launch(void* const* d_in, const int* in_sizes, int n_in,
                              void* d_out, int out_size, void* d_ws, size_t ws_size,
                              hipStream_t stream) {
  (void)in_sizes; (void)n_in; (void)out_size; (void)d_ws; (void)ws_size;
  const float* logits = (const float*)d_in[0];
  const float* value  = (const float*)d_in[1];
  const float* sinks  = (const float*)d_in[2];
  float* out = (float*)d_out;

  dim3 grid(NH * (SQ / 64));  // 1024 workgroups: (head, 64-row q block)
  dim3 block(128);            // 4 waves; each wave: 16 rows x 128 DH
  hipLaunchKernelGGL(attn_sink_pv, grid, block, 0, stream, logits, value, sinks, out);
}
